// ExpandedTerrainFeatures_18751827214368
// MI455X (gfx1250) — compile-verified, matrix operand order checked by reading, not run on hardware
//
#include <hip/hip_runtime.h>
#include <math.h>

typedef __attribute__((ext_vector_type(16))) _Float16 v16h;
typedef __attribute__((ext_vector_type(8)))  float    v8f;

#define EPSF 1e-6f

// ---------------- wave32 reduction helpers ----------------
__device__ __forceinline__ float wsum(float v) {
#pragma unroll
  for (int m = 1; m < 32; m <<= 1) v += __shfl_xor(v, m, 32);
  return v;
}
__device__ __forceinline__ float wmaxr(float v) {
#pragma unroll
  for (int m = 1; m < 32; m <<= 1) v = fmaxf(v, __shfl_xor(v, m, 32));
  return v;
}

// ---------------- register bitonic sort of 256 (8 per lane, idx = lane*8+i) ----
__device__ __forceinline__ void cex(float& a, float& b, bool asc) {
  float lo = fminf(a, b), hi = fmaxf(a, b);
  a = asc ? lo : hi;
  b = asc ? hi : lo;
}
template <int J>
__device__ __forceinline__ void intraStage(float v[8], int lane, int k) {
#pragma unroll
  for (int i = 0; i < 8; i++) {
    if ((i & J) == 0) {
      bool asc = (((lane * 8 + i) & k) == 0);
      cex(v[i], v[i + J], asc);
    }
  }
}
__device__ __forceinline__ void sort256(float v[8], int lane) {
  for (int k = 2; k <= 256; k <<= 1) {
    for (int j = k >> 1; j >= 8; j >>= 1) {
      int lm = j >> 3;
      bool lower = ((lane & lm) == 0);
#pragma unroll
      for (int i = 0; i < 8; i++) {
        float o = __shfl_xor(v[i], lm, 32);
        bool asc = (((lane * 8 + i) & k) == 0);
        float mn = fminf(v[i], o), mx = fmaxf(v[i], o);
        v[i] = (lower == asc) ? mn : mx;
      }
    }
    if (k >= 8) intraStage<4>(v, lane, k);
    if (k >= 4) intraStage<2>(v, lane, k);
    intraStage<1>(v, lane, k);
  }
}
template <int N>
__device__ __forceinline__ float sortedVal(const float v[8]) {
  return __shfl(v[N & 7], N >> 3, 32);
}

// ---------------- per-signal summary (signals are norms, >= 0) ----------------
__device__ __forceinline__ void summaryTask(const float* sp, float* s_out,
                                            float* s_rms, float* s_max, int s, int lane) {
  float x[8];
#pragma unroll
  for (int i = 0; i < 8; i++) x[i] = sp[lane * 8 + i];
  float s1a = 0.f, s2 = 0.f, mx = 0.f, sx = 0.f;
#pragma unroll
  for (int i = 0; i < 8; i++) {
    float a = fabsf(x[i]);
    s1a += a; s2 += x[i] * x[i]; mx = fmaxf(mx, a); sx += x[i];
  }
  s1a = wsum(s1a); s2 = wsum(s2); mx = wmaxr(mx); sx = wsum(sx);
  float mean = sx * (1.0f / 256.0f);
  float c2 = 0.f, c3 = 0.f, c4 = 0.f;
#pragma unroll
  for (int i = 0; i < 8; i++) {
    float d = x[i] - mean, d2 = d * d;
    c2 += d2; c3 += d2 * d; c4 += d2 * d2;
  }
  c2 = wsum(c2); c3 = wsum(c3); c4 = wsum(c4);
  sort256(x, lane);  // x >= 0, so sorted(x) == sorted(|x|)
  float S63 = sortedVal<63>(x), S64 = sortedVal<64>(x);
  float S191 = sortedVal<191>(x), S192 = sortedVal<192>(x);
  float S242 = sortedVal<242>(x), S243 = sortedVal<243>(x);
  if (lane == 0) {
    float rms = sqrtf(s2 * (1.0f / 256.0f));
    float var = fmaxf(c2 * (1.0f / 256.0f), EPSF);
    float skew = (c3 * (1.0f / 256.0f)) / (var * sqrtf(var));
    skew = fminf(fmaxf(skew, -10.0f), 10.0f);
    float kurt = (c4 * (1.0f / 256.0f)) / (var * var);
    kurt = fminf(fmaxf(kurt, 0.0f), 30.0f);
    float q95 = S242 + 0.25f * (S243 - S242);
    float q75 = S191 + 0.25f * (S192 - S191);
    float q25 = S63 + 0.75f * (S64 - S63);
    float* o = s_out + s * 8;
    o[0] = s1a * (1.0f / 256.0f);
    o[1] = sqrtf(c2 * (1.0f / 255.0f));
    o[2] = rms;
    o[3] = mx;
    o[4] = q95;
    o[5] = q75 - q25;
    o[6] = skew;
    o[7] = kurt;
    s_rms[s] = rms;
    s_max[s] = mx;
  }
}

// ---------------- heel/toe phase features ----------------
__device__ __forceinline__ void phaseTask(const float* zp, float* outp, int sT, int R, int lane) {
  float bv = -1.0f; int bi = 0;
  for (int t = lane; t < sT; t += 32) {
    float a = fabsf(zp[t]);
    if (a > bv) { bv = a; bi = t; }
  }
#pragma unroll
  for (int m = 1; m < 32; m <<= 1) {
    float ov = __shfl_xor(bv, m, 32);
    int   oi = __shfl_xor(bi, m, 32);
    if (ov > bv || (ov == bv && oi < bi)) { bv = ov; bi = oi; }
  }
  float pk = bv;
  int W = 2 * R + 1;
  float pre = 0.f, post = 0.f, tot = 0.f, cnt = 0.f;
  for (int j = lane; j < W; j += 32) {
    int t = bi - R + j;
    t = t < 0 ? 0 : (t > sT - 1 ? sT - 1 : t);
    float val = fabsf(zp[t]);
    tot += val;
    if (j < R) pre += val;
    if (j > R) post += val;
    if (val >= pk * 0.2f) cnt += 1.0f;
  }
  pre = wsum(pre); post = wsum(post); tot = wsum(tot); cnt = wsum(cnt);
  float jm = 0.f, js = 0.f;
  for (int t = lane; t < sT - 1; t += 32) {
    float d = zp[t + 1] - zp[t];
    jm = fmaxf(jm, fabsf(d)); js += d * d;
  }
  jm = wmaxr(jm); js = wsum(js);
  if (lane == 0) {
    outp[0] = pk;
    outp[1] = tot;
    outp[2] = (post / (float)R) / ((pre / (float)R) + EPSF);
    outp[3] = cnt / (float)W;
    outp[4] = jm;
    outp[5] = sqrtf(js / (float)(sT - 1));
  }
}

// ---------------- time-domain xcorr, lags -8..8 (== zero-padded FFT corr) -----
__device__ __forceinline__ void xcorrTask(const float* xp, const float* yp,
                                          float* s_out, float* s_zmax, int i, int lane) {
  float sx = 0.f, sy = 0.f, ax = 0.f, ay = 0.f;
#pragma unroll
  for (int n = 0; n < 8; n++) {
    int t = lane + 32 * n;
    float xv = xp[t], yv = yp[t];
    sx += xv; sy += yv;
    ax = fmaxf(ax, fabsf(xv)); ay = fmaxf(ay, fabsf(yv));
  }
  sx = wsum(sx); sy = wsum(sy); ax = wmaxr(ax); ay = wmaxr(ay);
  float mx = sx * (1.0f / 256.0f), my = sy * (1.0f / 256.0f);
  float x0[8]; float nx = 0.f, ny = 0.f;
#pragma unroll
  for (int n = 0; n < 8; n++) {
    int t = lane + 32 * n;
    x0[n] = xp[t] - mx; nx += x0[n] * x0[n];
    float y0 = yp[t] - my; ny += y0 * y0;
  }
  nx = wsum(nx); ny = wsum(ny);
  float best = -1e30f; int bj = 0;
  for (int j = 0; j < 17; j++) {
    int d = j - 8;
    float acc = 0.f;
#pragma unroll
    for (int n = 0; n < 8; n++) {
      int t = lane + 32 * n;
      int u = t - d;
      if (u >= 0 && u < 256) acc += x0[n] * (yp[u] - my);
    }
    acc = wsum(acc);
    if (acc > best) { best = acc; bj = j; }
  }
  float denom = sqrtf(nx) * sqrtf(ny) + EPSF;
  if (lane == 0) {
    int base = 172 + 6 * i;
    s_out[base + 3] = best / denom;
    s_out[base + 4] = (float)(bj - 8);
    s_zmax[i] = ax;
    s_zmax[2 + i] = ay;
  }
}

// ---------------- horizontal (xy) features ----------------
__device__ __forceinline__ void horizTask(const float* xc, const float* yc, const float* zc,
                                          float* outp, int lane) {
  float h[8]; float sh2 = 0.f, hm = 0.f, az = 0.f;
#pragma unroll
  for (int i = 0; i < 8; i++) {
    int t = lane * 8 + i;
    float a = xc[t], b = yc[t];
    h[i] = sqrtf(a * a + b * b);
    sh2 += h[i] * h[i];
    hm = fmaxf(hm, h[i]);
    az += fabsf(zc[t]);
  }
  sh2 = wsum(sh2); hm = wmaxr(hm); az = wsum(az);
  float jm = 0.f, js = 0.f;
  for (int t = lane; t < 255; t += 32) {
    float a0 = xc[t], b0 = yc[t], a1 = xc[t + 1], b1 = yc[t + 1];
    float d = sqrtf(a1 * a1 + b1 * b1) - sqrtf(a0 * a0 + b0 * b0);
    jm = fmaxf(jm, fabsf(d)); js += d * d;
  }
  jm = wmaxr(jm); js = wsum(js);
  sort256(h, lane);
  float S242 = sortedVal<242>(h), S243 = sortedVal<243>(h);
  if (lane == 0) {
    float r = sqrtf(sh2 * (1.0f / 256.0f));
    outp[0] = r;
    outp[1] = hm;
    outp[2] = S242 + 0.25f * (S243 - S242);
    outp[3] = jm;
    outp[4] = sqrtf(js * (1.0f / 255.0f));
    outp[5] = r / (az * (1.0f / 256.0f) + EPSF);
  }
}

// ---------------- spectral post-processing (bands, entropy, SEF85) ------------
__device__ __forceinline__ void spectralPost(const float* C, const float* S,
                                             float* s_out, int sig, int lane) {
  float b0 = 0, b1 = 0, b2 = 0, b3 = 0, b4 = 0, tot = 0;
  for (int k = lane; k < 129; k += 32) {
    float c = C[k], s = S[k];
    float P = c * c + s * s;
    tot += P;
    float fq = (float)k * 0.390625f;
    if (fq < 3.0f) b0 += P;
    else if (fq < 6.0f) b1 += P;
    else if (fq < 10.0f) b2 += P;
    else if (fq < 20.0f) b3 += P;
    else if (fq < 40.0f) b4 += P;
  }
  tot = wsum(tot); b0 = wsum(b0); b1 = wsum(b1); b2 = wsum(b2); b3 = wsum(b3); b4 = wsum(b4);
  float total = fmaxf(tot, 1e-8f);
  float ent = 0.f;
  for (int k = lane; k < 129; k += 32) {
    float c = C[k], s = S[k];
    float P = c * c + s * s;
    float p = fmaxf(P / total, 1e-8f);
    ent += p * logf(p);
  }
  ent = wsum(ent);
  if (lane == 0) {
    float* o = s_out + 96 + sig * 7;
    o[0] = b0 / total; o[1] = b1 / total; o[2] = b2 / total;
    o[3] = b3 / total; o[4] = b4 / total;
    o[5] = -ent / 4.867534450455582f;  // log(130)
    float csum = 0.f; int kf = 0; bool fnd = false;
    for (int k = 0; k < 129 && !fnd; k++) {
      float c = C[k], s = S[k];
      float P = c * c + s * s;
      csum += fmaxf(P / total, 1e-8f);
      if (csum >= 0.85f) { kf = k; fnd = true; }
    }
    o[6] = (float)kf * 0.390625f;
  }
}

// =====================================================================
// Fused kernel: one workgroup (8 wave32) per batch row.
// =====================================================================
__global__ __launch_bounds__(256) void gait_features_kernel(
    const float* __restrict__ foot, const float* __restrict__ shank,
    const float* __restrict__ thigh, float* __restrict__ out) {
  __shared__ float s_raw[3][3072];      // foot/shank/thigh rows (12ch x 256)
  __shared__ float s_sig[12 * 256];     // 12 norm signals
  __shared__ float s_C[4][132];         // DFT cos coefficients per z-signal
  __shared__ float s_S[4][132];         // DFT sin coefficients per z-signal
  __shared__ float s_outF[208];         // staged feature vector
  __shared__ float s_rms[12], s_max[12], s_zmax[4];
  __shared__ _Float16 s_trig[256];      // sin(2*pi*k/256) in f16 (cos = +64)

  const int tid  = threadIdx.x;
  const int lane = tid & 31;
  const int wid  = tid >> 5;
  const long long b = blockIdx.x;

  // ---- Phase 0: stage inputs (3 x 12KB) into LDS + build f16 sine table ----
  {
    const float4* g0 = (const float4*)(foot  + b * 3072);
    const float4* g1 = (const float4*)(shank + b * 3072);
    const float4* g2 = (const float4*)(thigh + b * 3072);
    float4* l0 = (float4*)&s_raw[0][0];
    float4* l1 = (float4*)&s_raw[1][0];
    float4* l2 = (float4*)&s_raw[2][0];
#pragma unroll
    for (int n = 0; n < 3; n++) {
      int idx = tid + n * 256;
      l0[idx] = g0[idx];
      l1[idx] = g1[idx];
      l2[idx] = g2[idx];
    }
    s_trig[tid] = (_Float16)__sinf((float)tid * 0.024543692606170258f);
  }
  __syncthreads();

  // ---- Phase 1: 12 norm signals (fa/fg/sa/sg/ta/tg, lt/rt) ----
  {
#pragma unroll
    for (int n = 0; n < 12; n++) {
      int e = tid + n * 256;
      int s = e >> 8, t = e & 255;
      int seg = s >> 2, j = s & 3;
      int cb = (j == 0) ? 0 : (j == 1) ? 6 : (j == 2) ? 3 : 9;
      const float* sp = &s_raw[seg][0];
      float a = sp[cb * 256 + t], bb = sp[(cb + 1) * 256 + t], c = sp[(cb + 2) * 256 + t];
      s_sig[s * 256 + t] = sqrtf(a * a + bb * bb + c * c);
    }
  }
  __syncthreads();

  // ---- Phase 2: DFT power spectrum via v_wmma_f32_16x16x32_f16 ----
  // A (16x256 f16): rows 0-3 = f16-hi of z-signals, rows 4-7 = f16 residual.
  // Rows 8-15 land only in lanes 16-31 of D, which are never read, so they are
  // left as (harmless) duplicates -> A build is fully branchless.
  // B tiles come from the LDS f16 sine table: elem = sin[(kf*t + 64*isCos)&255].
  // Out-of-range columns (tile 16) only pollute their own never-read D lanes.
  {
    const int m = lane & 15;
    const int khalf = (lane & 16) ? 8 : 0;
    const int sg = m & 3;
    const float* zp = &s_raw[(sg >= 2) ? 1 : 0][((sg & 1) ? 8 : 2) * 256];
    v16h afrag[8];
#pragma unroll
    for (int kk = 0; kk < 8; kk++) {
      v16h a;
#pragma unroll
      for (int u = 0; u < 16; u++) {
        int t = kk * 32 + khalf + ((u < 8) ? u : (u + 8));
        float xv = zp[t];
        _Float16 hi = (_Float16)xv;
        a[u] = (m < 4) ? hi : (_Float16)(xv - (float)hi);
      }
      afrag[kk] = a;
    }
    for (int nt = wid; nt < 17; nt += 8) {
      int col = nt * 16 + (lane & 15);
      bool isCos = (col < 129);
      bool valid = (col < 258);
      int kf = isCos ? col : (col - 129);
      int cosoff = isCos ? 64 : 0;
      v8f acc = {};
#pragma unroll
      for (int kk = 0; kk < 8; kk++) {
        v16h bfrag;
#pragma unroll
        for (int u = 0; u < 16; u++) {
          int t = kk * 32 + khalf + ((u < 8) ? u : (u + 8));
          int idx = (kf * t + cosoff) & 255;
          bfrag[u] = s_trig[idx];
        }
        acc = __builtin_amdgcn_wmma_f32_16x16x32_f16(
            false, afrag[kk], false, bfrag, (short)0, acc, false, false);
      }
      if (lane < 16 && valid) {
#pragma unroll
        for (int r = 0; r < 4; r++) {
          float coef = acc[r] + acc[r + 4];
          if (isCos) s_C[r][col] = coef;
          else       s_S[r][col - 129] = coef;
        }
      }
    }
  }
  __syncthreads();

  // ---- Phase 3: per-wave task mix ----
  if (wid < 4) {
    spectralPost(s_C[wid], s_S[wid], s_outF, wid, lane);
    summaryTask(&s_sig[wid * 256], s_outF, s_rms, s_max, wid, lane);
    {
      int sg = wid;
      const float* zp = &s_raw[(sg >= 2) ? 1 : 0][((sg & 1) ? 8 : 2) * 256];
      phaseTask(zp, s_outF + 124 + sg * 6, 115, 19, lane);  // heel
    }
    if (wid == 0)      xcorrTask(&s_raw[0][2 * 256], &s_raw[1][2 * 256], s_outF, s_zmax, 0, lane);
    else if (wid == 1) xcorrTask(&s_raw[0][8 * 256], &s_raw[1][8 * 256], s_outF, s_zmax, 1, lane);
    else if (wid == 2) horizTask(&s_raw[0][0],       &s_raw[0][256],     &s_raw[0][512],     s_outF + 184, lane);
    else               horizTask(&s_raw[0][6 * 256], &s_raw[0][7 * 256], &s_raw[0][8 * 256], s_outF + 190, lane);
  } else {
    summaryTask(&s_sig[wid * 256], s_outF, s_rms, s_max, wid, lane);
    summaryTask(&s_sig[(wid + 4) * 256], s_outF, s_rms, s_max, wid + 4, lane);
    int sg = wid - 4;
    const float* zp = &s_raw[(sg >= 2) ? 1 : 0][((sg & 1) ? 8 : 2) * 256] + 153;
    phaseTask(zp, s_outF + 148 + sg * 6, 103, 17, lane);    // toe
  }
  __syncthreads();

  // ---- Phase 4: cross features (blocks 5/6 rest + asymmetry block 9) ----
  if (tid < 12) {
    float a, bv;
    int t = tid;
    if (t < 8) {
      int p = t >> 1;
      int sA = 2 * p, sB = 2 * p + 1;
      a  = (t & 1) ? s_rms[sA] : s_max[sA];
      bv = (t & 1) ? s_rms[sB] : s_max[sB];
    } else if (t == 8)  { a = s_rms[8];  bv = s_rms[9];  }
    else if (t == 9)    { a = s_rms[10]; bv = s_rms[11]; }
    else if (t == 10)   { a = s_outF[124 + 1];      bv = s_outF[124 + 6 + 1];  }
    else                { a = s_outF[124 + 12 + 1]; bv = s_outF[124 + 18 + 1]; }
    s_outF[196 + t] = fabsf(logf((a + EPSF) / (bv + EPSF)));
  } else if (tid < 24) {
    int l = tid - 12;
    int i = l / 6, k = l % 6;
    int base = 172 + 6 * i;
    float ratio = s_rms[4 + i] / (s_rms[i] + EPSF);
    if (k == 0)      s_outF[base]     = s_zmax[2 + i] / (s_zmax[i] + EPSF);
    else if (k == 1) s_outF[base + 1] = ratio;
    else if (k == 2) s_outF[base + 2] = s_outF[124 + (2 + i) * 6 + 1] /
                                        (s_outF[124 + i * 6 + 1] + EPSF);
    else if (k == 5) s_outF[base + 5] = 0.5f * (s_outF[96 + (2 + i) * 7 + 4] /
                                        (s_outF[96 + i * 7 + 4] + EPSF) + (1.0f - ratio));
    // k==3,4 written by xcorrTask
  }
  __syncthreads();

  // ---- Phase 5: coalesced store of the 208-feature row ----
  if (tid < 208) out[b * 208 + tid] = s_outF[tid];
}

extern "C" void kernel_launch(void* const* d_in, const int* in_sizes, int n_in,
                              void* d_out, int out_size, void* d_ws, size_t ws_size,
                              hipStream_t stream) {
  const float* foot  = (const float*)d_in[0];
  const float* shank = (const float*)d_in[1];
  const float* thigh = (const float*)d_in[2];
  float* out = (float*)d_out;
  int B = in_sizes[0] / 3072;  // 12 channels * 256 samples
  dim3 grid(B), block(256);
  hipLaunchKernelGGL(gait_features_kernel, grid, block, 0, stream,
                     foot, shank, thigh, out);
  (void)n_in; (void)out_size; (void)d_ws; (void)ws_size;
}
